// MultiHeadLatentAttention_66795331387843
// MI455X (gfx1250) — matrix operand order repaired and checked by reading, not compiled, not run on hardware
//
#include <hip/hip_runtime.h>
#include <hip/hip_bf16.h>

#define S_LEN 2048
#define BATCH 2
#define HIDD  2048
#define NHEAD 16
#define DQN_  128
#define DR_   64
#define DV_   128
#define QLR_  1536
#define KVLR_ 512
#define DQ_   192
#define NTOK  (S_LEN*BATCH)

typedef __bf16 bf16;
typedef __attribute__((ext_vector_type(16))) __bf16       v16bf;
typedef __attribute__((ext_vector_type(4)))  __bf16       v4bf;
typedef __attribute__((ext_vector_type(8)))  float        v8f;
typedef __attribute__((ext_vector_type(4)))  unsigned int v4u;

union FragU { v16bf f; v4u u[2]; };

__device__ __forceinline__ v8f zero8() {
  v8f z;
#pragma unroll
  for (int i = 0; i < 8; ++i) z[i] = 0.0f;
  return z;
}

__device__ __forceinline__ v8f wmma_bf16(v16bf a, v16bf b, v8f c) {
  return __builtin_amdgcn_wmma_f32_16x16x32_bf16(false, a, false, b, (short)0, c, false, false);
}

// A-matrix fragment (16x32, M x K), row-major source, ld in elements.
// lane<16: row=lane,    K=0..15
// lane>=16: row=lane-16, K=16..31
__device__ __forceinline__ v16bf load_fragA(const bf16* __restrict__ p, int ld) {
  int lane = threadIdx.x & 31;
  const bf16* q = p + (size_t)(lane & 15) * ld + ((lane >> 4) << 4);
  FragU x;
  x.u[0] = *reinterpret_cast<const v4u*>(q);
  x.u[1] = *reinterpret_cast<const v4u*>(q + 8);
  return x.f;
}

// B-matrix fragment (32x16, K x N) sourced from row-major N x K storage
// (i.e. we compute A*W^T; row n of the source supplies column n of B).
// lane<16: col=lane, K=0..15 ; lane>=16: col=lane-16, K=16..31
__device__ __forceinline__ v16bf load_fragB(const bf16* __restrict__ p, int ld) {
  int lane = threadIdx.x & 31;
  const bf16* q = p + (size_t)(lane & 15) * ld + ((lane >> 4) << 4);
  FragU x;
  x.u[0] = *reinterpret_cast<const v4u*>(q);
  x.u[1] = *reinterpret_cast<const v4u*>(q + 8);
  return x.f;
}

__device__ __forceinline__ v16bf load_fragB_guard(const bf16* __restrict__ base,
                                                  int n0, int N, int K, int k0) {
  int lane = threadIdx.x & 31;
  int row = n0 + (lane & 15);
  FragU x;
  if (row < N) {
    const bf16* q = base + (size_t)row * K + k0 + ((lane >> 4) << 4);
    x.u[0] = *reinterpret_cast<const v4u*>(q);
    x.u[1] = *reinterpret_cast<const v4u*>(q + 8);
  } else {
#pragma unroll
    for (int i = 0; i < 16; ++i) x.f[i] = (bf16)0.0f;
  }
  return x.f;
}

// ---------------- elementwise fp32 -> bf16 (4 elems / thread) ----------------
__global__ void mla_cvt_bf16(const float4* __restrict__ x, v4bf* __restrict__ y, long n4) {
  long i = (long)blockIdx.x * blockDim.x + threadIdx.x;
  if (i < n4) {
    float4 v = x[i];
    v4bf o;
    o[0] = (bf16)v.x; o[1] = (bf16)v.y; o[2] = (bf16)v.z; o[3] = (bf16)v.w;
    y[i] = o;
  }
}

// ---------------- generic GEMM: C[M,N] = A[M,K] * W[N,K]^T ----------------
// block: 256 threads = 8 waves (4 along M x 2 along N), wave tile 32x64,
// block tile 128x128.  Interior blocks take a guard-free inner loop.
__global__ __launch_bounds__(256)
void mla_gemm_xwt(const bf16* __restrict__ A, const bf16* __restrict__ W,
                  float* __restrict__ C, int M, int N, int K) {
  int wave = threadIdx.x >> 5;
  int m0 = blockIdx.y * 128 + (wave & 3) * 32;
  int n0 = blockIdx.x * 128 + (wave >> 2) * 64;
  v8f acc[2][4];
#pragma unroll
  for (int mi = 0; mi < 2; ++mi)
#pragma unroll
    for (int ni = 0; ni < 4; ++ni) acc[mi][ni] = zero8();

  const bf16* a0p = A + (size_t)m0 * K;
  const bf16* a1p = A + (size_t)(m0 + 16) * K;
  const bool full = (blockIdx.x * 128 + 128) <= (unsigned)N;

  if (full) {
    for (int k = 0; k < K; k += 32) {
      v16bf a0 = load_fragA(a0p + k, K);
      v16bf a1 = load_fragA(a1p + k, K);
      v16bf b[4];
#pragma unroll
      for (int ni = 0; ni < 4; ++ni)
        b[ni] = load_fragB(W + (size_t)(n0 + 16 * ni) * K + k, K);
#pragma unroll
      for (int ni = 0; ni < 4; ++ni) {
        acc[0][ni] = wmma_bf16(a0, b[ni], acc[0][ni]);
        acc[1][ni] = wmma_bf16(a1, b[ni], acc[1][ni]);
      }
    }
  } else {
    for (int k = 0; k < K; k += 32) {
      v16bf a0 = load_fragA(a0p + k, K);
      v16bf a1 = load_fragA(a1p + k, K);
      v16bf b[4];
#pragma unroll
      for (int ni = 0; ni < 4; ++ni)
        b[ni] = load_fragB_guard(W, n0 + 16 * ni, N, K, k);
#pragma unroll
      for (int ni = 0; ni < 4; ++ni) {
        acc[0][ni] = wmma_bf16(a0, b[ni], acc[0][ni]);
        acc[1][ni] = wmma_bf16(a1, b[ni], acc[1][ni]);
      }
    }
  }

  int lane = threadIdx.x & 31;
  int nn = lane & 15, mb = (lane >> 4) << 3;
  if (full) {
#pragma unroll
    for (int mi = 0; mi < 2; ++mi)
#pragma unroll
      for (int r = 0; r < 8; ++r) {
        size_t m = m0 + 16 * mi + mb + r;
#pragma unroll
        for (int ni = 0; ni < 4; ++ni)
          C[m * N + n0 + 16 * ni + nn] = acc[mi][ni][r];
      }
  } else {
#pragma unroll
    for (int mi = 0; mi < 2; ++mi)
#pragma unroll
      for (int r = 0; r < 8; ++r) {
        size_t m = m0 + 16 * mi + mb + r;
#pragma unroll
        for (int ni = 0; ni < 4; ++ni) {
          int nc = n0 + 16 * ni + nn;
          if (nc < N) C[m * N + nc] = acc[mi][ni][r];
        }
      }
  }
}

// ---------------- RMSNorm: Y(bf16)[row,cols] = x*rsqrt(mean(x^2)+eps)*w ----------------
__global__ __launch_bounds__(256)
void mla_rmsnorm(const float* __restrict__ X, const float* __restrict__ w,
                 bf16* __restrict__ Y, int cols, int in_stride) {
  int row = blockIdx.x;
  const float* x = X + (size_t)row * in_stride;
  float s = 0.0f;
  for (int i = threadIdx.x; i < cols; i += 256) { float v = x[i]; s += v * v; }
  __shared__ float red[256];
  red[threadIdx.x] = s;
  __syncthreads();
  for (int off = 128; off > 0; off >>= 1) {
    if (threadIdx.x < off) red[threadIdx.x] += red[threadIdx.x + off];
    __syncthreads();
  }
  float r = rsqrtf(red[0] / (float)cols + 1e-6f);
  for (int i = threadIdx.x; i < cols; i += 256)
    Y[(size_t)row * cols + i] = (bf16)((x[i] * r) * w[i]);
}

// ---------------- build Q with RoPE: Q[b][h][s][DQ] bf16 ----------------
__global__ void mla_build_q(const float* __restrict__ qraw, const int* __restrict__ pos_ids,
                            bf16* __restrict__ Q) {
  int t = blockIdx.x, h = blockIdx.y;
  int s = t / BATCH, b = t % BATCH;
  const float* src = qraw + (size_t)t * (NHEAD * DQ_) + h * DQ_;
  bf16* dst = Q + (((size_t)(b * NHEAD + h)) * S_LEN + s) * DQ_;
  int tid = threadIdx.x;
  if (tid < DQN_) {
    dst[tid] = (bf16)src[tid];
  } else {
    int j = tid - DQN_;               // 0..63
    int i = j & 31;
    int pos = pos_ids[b * S_LEN + s];
    float f = __powf(10000.0f, -(float)i / 32.0f);
    float ang = (float)pos * f;
    float c = __cosf(ang), sn = __sinf(ang);
    float e = src[DQN_ + 2 * i], o = src[DQN_ + 2 * i + 1];
    dst[tid] = (bf16)((j < 32) ? (e * c - o * sn) : (o * c + e * sn));
  }
}

// ---------------- build K (k_nope|rope(k_pe)) and V^T ----------------
__global__ void mla_build_kv(const float* __restrict__ kv, const float* __restrict__ ckv,
                             const int* __restrict__ pos_ids,
                             bf16* __restrict__ Kd, bf16* __restrict__ Vt) {
  int t = blockIdx.x, h = blockIdx.y;
  int s = t / BATCH, b = t % BATCH;
  const float* src = kv + (size_t)t * (NHEAD * (DQN_ + DV_)) + h * (DQN_ + DV_);
  bf16* kd = Kd + (((size_t)(b * NHEAD + h)) * S_LEN + s) * DQ_;
  int tid = threadIdx.x;
  if (tid < DQN_) {
    kd[tid] = (bf16)src[tid];
    Vt[(((size_t)(b * NHEAD + h)) * DV_ + tid) * S_LEN + s] = (bf16)src[DQN_ + tid];
  } else {
    int j = tid - DQN_;
    int i = j & 31;
    int pos = pos_ids[b * S_LEN + s];
    float f = __powf(10000.0f, -(float)i / 32.0f);
    float ang = (float)pos * f;
    float c = __cosf(ang), sn = __sinf(ang);
    const float* pe = ckv + (size_t)t * (KVLR_ + DR_) + KVLR_;
    float e = pe[2 * i], o = pe[2 * i + 1];
    kd[tid] = (bf16)((j < 32) ? (e * c - o * sn) : (o * c + e * sn));
  }
}

// ---------------- flash attention: 1 wave = 16 queries of one (b,h) ----------------
__global__ __launch_bounds__(32)
void mla_attn(const bf16* __restrict__ Q, const bf16* __restrict__ Kd,
              const bf16* __restrict__ Vt, bf16* __restrict__ O) {
  int m0 = blockIdx.x * 16;
  int h = blockIdx.y, b = blockIdx.z;
  int lane = threadIdx.x;
  const bf16* qp    = Q  + (((size_t)(b * NHEAD + h)) * S_LEN + m0) * DQ_;
  const bf16* kbase = Kd + ((size_t)(b * NHEAD + h)) * S_LEN * DQ_;
  const bf16* vbase = Vt + ((size_t)(b * NHEAD + h)) * DV_ * S_LEN;
  __shared__ __align__(16) bf16 pbuf[16 * 32];

  v16bf qf[6];
#pragma unroll
  for (int i = 0; i < 6; ++i) qf[i] = load_fragA(qp + 32 * i, DQ_);

  v8f o[8];
#pragma unroll
  for (int j = 0; j < 8; ++j) o[j] = zero8();
  float mrow[8], lrow[8];
#pragma unroll
  for (int r = 0; r < 8; ++r) { mrow[r] = -3.0e38f; lrow[r] = 0.0f; }

  const float scale = 0.0721687836f;  // 192^-0.5
  int nn = lane & 15;
  int mb = (lane >> 4) << 3;
  int kmax = m0 + 16;                 // causal: keys < m0+16 may be live

  for (int kt = 0; kt < kmax; kt += 32) {
    v8f s0 = zero8(), s1 = zero8();
#pragma unroll
    for (int i = 0; i < 6; ++i) {
      v16bf kf0 = load_fragB(kbase + (size_t)kt * DQ_ + 32 * i, DQ_);
      v16bf kf1 = load_fragB(kbase + (size_t)(kt + 16) * DQ_ + 32 * i, DQ_);
      s0 = wmma_bf16(qf[i], kf0, s0);
      s1 = wmma_bf16(qf[i], kf1, s1);
    }
    float alpha[8];
#pragma unroll
    for (int r = 0; r < 8; ++r) {
      int mg = m0 + mb + r;
      float a  = s0[r] * scale + (((kt + nn)      <= mg) ? 0.0f : -1.0e9f);
      float bb = s1[r] * scale + (((kt + 16 + nn) <= mg) ? 0.0f : -1.0e9f);
      float mx = fmaxf(a, bb);
      for (int off = 1; off < 16; off <<= 1) mx = fmaxf(mx, __shfl_xor(mx, off, 32));
      float mnew = fmaxf(mrow[r], mx);
      alpha[r] = __expf(mrow[r] - mnew);
      float p0 = __expf(a - mnew);
      float p1 = __expf(bb - mnew);
      s0[r] = p0; s1[r] = p1;
      float rsum = p0 + p1;
      for (int off = 1; off < 16; off <<= 1) rsum += __shfl_xor(rsum, off, 32);
      lrow[r] = lrow[r] * alpha[r] + rsum;
      mrow[r] = mnew;
    }
#pragma unroll
    for (int j = 0; j < 8; ++j)
#pragma unroll
      for (int r = 0; r < 8; ++r) o[j][r] *= alpha[r];

    // stage P (16x32) to LDS, reload in A-fragment layout
#pragma unroll
    for (int r = 0; r < 8; ++r) {
      pbuf[(mb + r) * 32 + nn]      = (bf16)s0[r];
      pbuf[(mb + r) * 32 + 16 + nn] = (bf16)s1[r];
    }
    asm volatile("s_wait_dscnt 0x0" ::: "memory");
    v16bf pf = load_fragA(pbuf, 32);
#pragma unroll
    for (int j = 0; j < 8; ++j) {
      v16bf vf = load_fragB(vbase + (size_t)(16 * j) * S_LEN + kt, S_LEN);
      o[j] = wmma_bf16(pf, vf, o[j]);
    }
  }

#pragma unroll
  for (int r = 0; r < 8; ++r) {
    int sg = m0 + mb + r;
    size_t t = (size_t)sg * BATCH + b;
    float inv = 1.0f / lrow[r];
#pragma unroll
    for (int j = 0; j < 8; ++j)
      O[t * (NHEAD * DV_) + h * DV_ + 16 * j + nn] = (bf16)(o[j][r] * inv);
  }
}

// ======================= host side =======================
static inline unsigned cdiv(long a, long b) { return (unsigned)((a + b - 1) / b); }

static inline void launch_cvt(const float* src, bf16* dst, long n, hipStream_t stream) {
  long n4 = n / 4;  // all sizes here are multiples of 4
  mla_cvt_bf16<<<cdiv(n4, 256), 256, 0, stream>>>(
      (const float4*)src, (v4bf*)dst, n4);
}

extern "C" void kernel_launch(void* const* d_in, const int* in_sizes, int n_in,
                              void* d_out, int out_size, void* d_ws, size_t ws_size,
                              hipStream_t stream) {
  const float* hidden    = (const float*)d_in[0];
  // d_in[1] attention_mask: exact causal 0/-1e9 mask, applied analytically in-kernel
  const int*   pos_ids   = (const int*)d_in[2];
  const float* wq_a      = (const float*)d_in[3];
  const float* q_a_ln_w  = (const float*)d_in[4];
  const float* wq_b      = (const float*)d_in[5];
  const float* wkv_a     = (const float*)d_in[6];
  const float* kv_a_ln_w = (const float*)d_in[7];
  const float* wkv_b     = (const float*)d_in[8];
  const float* wo        = (const float*)d_in[9];
  float* out = (float*)d_out;

  // ---- workspace arena (regions reused once their producer/consumer chain ends) ----
  char* base = (char*)d_ws;
  size_t off = 0;
  auto nxt = [&](size_t bytes) -> char* {
    char* p = base + off;
    off += (bytes + 255) & ~(size_t)255;
    return p;
  };
  bf16*  hiddenB = (bf16*)nxt((size_t)NTOK * HIDD * 2);
  bf16*  wbuf    = (bf16*)nxt((size_t)(NHEAD * DQ_) * QLR_ * 2);           // largest weight (wq_b)
  float* big1    = (float*)nxt((size_t)NTOK * (NHEAD * (DQN_ + DV_)) * 4); // q_lat -> kv
  float* big2    = (float*)nxt((size_t)NTOK * (NHEAD * DQ_) * 4);          // q_raw -> attnO
  char*  buf4    = nxt((size_t)NTOK * QLR_ * 2);                           // qnorm(bf16) -> ckv(f32)
  bf16*  ckvnorm = (bf16*)nxt((size_t)NTOK * KVLR_ * 2);
  bf16*  Qb      = (bf16*)nxt((size_t)BATCH * NHEAD * S_LEN * DQ_ * 2);
  bf16*  Kb      = (bf16*)nxt((size_t)BATCH * NHEAD * S_LEN * DQ_ * 2);
  bf16*  Vtb     = (bf16*)nxt((size_t)BATCH * NHEAD * DV_ * S_LEN * 2);
  (void)ws_size; (void)in_sizes; (void)n_in; (void)out_size;

  // hidden -> bf16
  launch_cvt(hidden, hiddenB, (long)NTOK * HIDD, stream);
  // G1: q_lat = hidden @ wq_a^T   (4096 x 1536)
  launch_cvt(wq_a, wbuf, (long)QLR_ * HIDD, stream);
  mla_gemm_xwt<<<dim3(cdiv(QLR_, 128), NTOK / 128), 256, 0, stream>>>(
      hiddenB, wbuf, big1, NTOK, QLR_, HIDD);
  // rmsnorm(q_lat) -> bf16
  mla_rmsnorm<<<NTOK, 256, 0, stream>>>(big1, q_a_ln_w, (bf16*)buf4, QLR_, QLR_);
  // G2: q_raw = qnorm @ wq_b^T   (4096 x 3072)
  launch_cvt(wq_b, wbuf, (long)(NHEAD * DQ_) * QLR_, stream);
  mla_gemm_xwt<<<dim3(cdiv(NHEAD * DQ_, 128), NTOK / 128), 256, 0, stream>>>(
      (const bf16*)buf4, wbuf, big2, NTOK, NHEAD * DQ_, QLR_);
  // G3: ckv = hidden @ wkv_a^T   (4096 x 576) -> overwrite buf4 as f32
  launch_cvt(wkv_a, wbuf, (long)(KVLR_ + DR_) * HIDD, stream);
  mla_gemm_xwt<<<dim3(cdiv(KVLR_ + DR_, 128), NTOK / 128), 256, 0, stream>>>(
      hiddenB, wbuf, (float*)buf4, NTOK, KVLR_ + DR_, HIDD);
  // rmsnorm(compressed_kv) -> bf16
  mla_rmsnorm<<<NTOK, 256, 0, stream>>>((const float*)buf4, kv_a_ln_w, ckvnorm,
                                        KVLR_, KVLR_ + DR_);
  // G4: kv = ckvnorm @ wkv_b^T   (4096 x 4096) -> big1 (q_lat dead)
  launch_cvt(wkv_b, wbuf, (long)(NHEAD * (DQN_ + DV_)) * KVLR_, stream);
  mla_gemm_xwt<<<dim3(cdiv(NHEAD * (DQN_ + DV_), 128), NTOK / 128), 256, 0, stream>>>(
      ckvnorm, wbuf, big1, NTOK, NHEAD * (DQN_ + DV_), KVLR_);
  // RoPE + layout builds
  mla_build_q<<<dim3(NTOK, NHEAD), 192, 0, stream>>>(big2, pos_ids, Qb);
  mla_build_kv<<<dim3(NTOK, NHEAD), 192, 0, stream>>>(big1, (const float*)buf4, pos_ids, Kb, Vtb);
  // flash attention -> attnO bf16 (reuse big2; q_raw dead)
  bf16* attnO = (bf16*)big2;
  mla_attn<<<dim3(S_LEN / 16, NHEAD, BATCH), 32, 0, stream>>>(Qb, Kb, Vtb, attnO);
  // G5: out = attnO @ wo^T  (fp32 to d_out)
  launch_cvt(wo, wbuf, (long)HIDD * (NHEAD * DV_), stream);
  mla_gemm_xwt<<<dim3(cdiv(HIDD, 128), NTOK / 128), 256, 0, stream>>>(
      attnO, wbuf, out, NTOK, HIDD, NHEAD * DV_);
}
